// Attention_33509334843900
// MI455X (gfx1250) — matrix operand order repaired and checked by reading, not compiled, not run on hardware
//
#include <hip/hip_runtime.h>
#include <math.h>

typedef __bf16 bf16_t;
typedef __attribute__((ext_vector_type(16))) __bf16 v16bf;
typedef __attribute__((ext_vector_type(8)))  __bf16 v8bf;
typedef __attribute__((ext_vector_type(8)))  float  v8f;
typedef __attribute__((ext_vector_type(4)))  unsigned int v4u;
typedef __attribute__((ext_vector_type(8)))  int v8i;
typedef __attribute__((ext_vector_type(4)))  int v4i;

// ---------------- problem constants ----------------
#define BATCH   128
#define NH      12
#define HD      64
#define HH      14
#define WW      14
#define LL      196          // 14*14
#define DIM     768
#define N3      2304         // 3*768
#define MROWS   25088        // BATCH*LL
#define BB      1536         // NH*BATCH
#define QT      13           // ceil(196/16)
#define KPAD    224          // keys padded to multiple of 32
#define LDS_STRIDE 72        // 64 elems + 8 pad elems (16B) -> conflict-free ds_load_b128

#if __has_builtin(__builtin_amdgcn_tensor_load_to_lds)
#define HAS_TDM 1
#else
#define HAS_TDM 0
#endif

// ---------------- WMMA helpers ----------------
__device__ inline v8f wmma_bf16(v16bf a, v16bf b, v8f c) {
    return __builtin_amdgcn_wmma_f32_16x16x32_bf16(
        false, a, false, b, (short)0, c, false, false);
}

// A fragment: 16x32 bf16 from row-major A (lda elements per row).
// lane: m = lane&15, half = lane>>4; elems i<8 -> K=half*8+i, i>=8 -> K=16+half*8+(i-8)
__device__ inline v16bf load_a(const bf16_t* A, int lda, int row0, int k0) {
    int lane = threadIdx.x & 31;
    int hf = lane >> 4;
    const bf16_t* p = A + (size_t)(row0 + (lane & 15)) * lda + k0 + hf * 8;
    v8bf c0 = *(const v8bf*)p;
    v8bf c1 = *(const v8bf*)(p + 16);
    v16bf out;
#pragma unroll
    for (int i = 0; i < 8; ++i) { out[i] = c0[i]; out[8 + i] = c1[i]; }
    return out;
}

__device__ inline v16bf load_a_guard(const bf16_t* A, int lda, int row0, int k0, int rlim) {
    int lane = threadIdx.x & 31;
    int hf = lane >> 4;
    int row = row0 + (lane & 15);
    v16bf out;
    if (row < rlim) {
        const bf16_t* p = A + (size_t)row * lda + k0 + hf * 8;
        v8bf c0 = *(const v8bf*)p;
        v8bf c1 = *(const v8bf*)(p + 16);
#pragma unroll
        for (int i = 0; i < 8; ++i) { out[i] = c0[i]; out[8 + i] = c1[i]; }
    } else {
#pragma unroll
        for (int i = 0; i < 16; ++i) out[i] = (bf16_t)0.0f;
    }
    return out;
}

// B fragment: 32x16 bf16; operand stored N-major: Bt[n][k].
// lane: n = lane&15, half = lane>>4; elems i<8 -> K=k0+half*8+i, i>=8 -> K=k0+16+half*8+(i-8)
__device__ inline v16bf load_b(const bf16_t* Bt, int ldb, int n0, int k0) {
    int lane = threadIdx.x & 31;
    int hf = lane >> 4;
    const bf16_t* p = Bt + (size_t)(n0 + (lane & 15)) * ldb + k0 + hf * 8;
    v8bf c0 = *(const v8bf*)p;
    v8bf c1 = *(const v8bf*)(p + 16);
    v16bf out;
#pragma unroll
    for (int i = 0; i < 8; ++i) { out[i] = c0[i]; out[8 + i] = c1[i]; }
    return out;
}

__device__ inline v16bf load_b_guard(const bf16_t* Bt, int ldb, int n0, int k0, int nlim) {
    int lane = threadIdx.x & 31;
    int hf = lane >> 4;
    int n = n0 + (lane & 15);
    v16bf out;
    if (n < nlim) {
        const bf16_t* p = Bt + (size_t)n * ldb + k0 + hf * 8;
        v8bf c0 = *(const v8bf*)p;
        v8bf c1 = *(const v8bf*)(p + 16);
#pragma unroll
        for (int i = 0; i < 8; ++i) { out[i] = c0[i]; out[8 + i] = c1[i]; }
    } else {
#pragma unroll
        for (int i = 0; i < 16; ++i) out[i] = (bf16_t)0.0f;
    }
    return out;
}

// LDS fragment loads, compile-time stride LDS_STRIDE
__device__ inline v16bf lds_load_a(const bf16_t* sA, int row0, int kk) {
    int lane = threadIdx.x & 31;
    int hf = lane >> 4;
    const bf16_t* p = sA + (row0 + (lane & 15)) * LDS_STRIDE + kk + hf * 8;
    v8bf c0 = *(const v8bf*)p;
    v8bf c1 = *(const v8bf*)(p + 16);
    v16bf out;
#pragma unroll
    for (int i = 0; i < 8; ++i) { out[i] = c0[i]; out[8 + i] = c1[i]; }
    return out;
}

__device__ inline v16bf lds_load_b(const bf16_t* sB, int n0, int kk) {
    int lane = threadIdx.x & 31;
    int hf = lane >> 4;
    const bf16_t* p = sB + (n0 + (lane & 15)) * LDS_STRIDE + kk + hf * 8;
    v8bf c0 = *(const v8bf*)p;
    v8bf c1 = *(const v8bf*)(p + 16);
    v16bf out;
#pragma unroll
    for (int i = 0; i < 8; ++i) { out[i] = c0[i]; out[8 + i] = c1[i]; }
    return out;
}

// ---------------- Tensor Data Mover: 128-row x 64-elem bf16 tile -> LDS ----------------
__device__ inline unsigned int lds_off(const void* p) {
    return (unsigned int)(uintptr_t)p;   // LDS aperture: low 32 bits = LDS byte offset
}

#if HAS_TDM
// D#: 2D tile, data_size=2B, tile_dim0=64 elems (128B = 32 DWORDs), tile_dim1=128 rows,
// pad_enable: +4 DWORDs (16B) after every 32 DWORDs -> LDS row stride 144B (LDS_STRIDE elems)
__device__ inline void tdm_load_tile(unsigned int lds_addr, const bf16_t* g, int ld_elems) {
    unsigned long long ga = (unsigned long long)(uintptr_t)g;
    v4u g0 = { 1u,                                  // count=1, user mode
               lds_addr,                            // lds_addr [63:32]
               (unsigned int)ga,                    // global_addr lo
               (unsigned int)(ga >> 32) | 0x80000000u };  // addr hi | type=2 ([127:126])
    const unsigned int td0 = 0x40000000u;           // huge tensor dims: no OOB clamping
    const unsigned int td1 = 0x40000000u;
    v8i g1 = {
        (int)((1u << 16)       // data_size = 2 bytes
            | (1u << 20)       // pad_enable
            | (4u << 22)       // pad_interval: 32 DWORDs
            | (3u << 25)),     // pad_amount: 4 DWORDs
        (int)((td0 & 0xFFFFu) << 16),                    // tensor_dim0 lo (atomic addr = 0)
        (int)((td0 >> 16) | ((td1 & 0xFFFFu) << 16)),    // tensor_dim0 hi | tensor_dim1 lo
        (int)((td1 >> 16) | (64u << 16)),                // tensor_dim1 hi | tile_dim0 = 64
        (int)128,                                        // tile_dim1 = 128, tile_dim2 = 0
        (int)ld_elems,                                   // tensor_dim0_stride lo
        0, 0 };                                          // stride hi / dim1 stride = 0
    v4i z4 = { 0, 0, 0, 0 };
#if defined(__clang_major__) && __clang_major__ >= 23
    v8i z8 = { 0, 0, 0, 0, 0, 0, 0, 0 };
    __builtin_amdgcn_tensor_load_to_lds(g0, g1, z4, z4, z8, 0);
#else
    __builtin_amdgcn_tensor_load_to_lds(g0, g1, z4, z4, 0);
#endif
}
#endif

// ---------------- shared 128x128 block GEMM core (8 waves, wave tile 32x64) ----------------
__device__ inline void compute_stage(const bf16_t* sA, const bf16_t* sB,
                                     int wm, int wn, v8f (&acc)[2][4]) {
#pragma unroll
    for (int kk = 0; kk < 64; kk += 32) {
        v16bf af[2], bfr[4];
#pragma unroll
        for (int mi = 0; mi < 2; ++mi) af[mi] = lds_load_a(sA, wm * 32 + mi * 16, kk);
#pragma unroll
        for (int ni = 0; ni < 4; ++ni) bfr[ni] = lds_load_b(sB, wn * 64 + ni * 16, kk);
#pragma unroll
        for (int mi = 0; mi < 2; ++mi)
#pragma unroll
            for (int ni = 0; ni < 4; ++ni)
                acc[mi][ni] = wmma_bf16(af[mi], bfr[ni], acc[mi][ni]);
    }
}

__device__ inline void gemm128_core(const bf16_t* __restrict__ A, const bf16_t* __restrict__ Bt,
                                    int K, int rowBase, int colBase, int lda, int ldb,
                                    v8f (&acc)[2][4]) {
    __shared__ bf16_t lA[2][128 * LDS_STRIDE];
    __shared__ bf16_t lB[2][128 * LDS_STRIDE];
    const int tid = (int)threadIdx.x;
    const int wid = tid >> 5;
    const int wm = wid >> 1, wn = wid & 1;
    const int nstage = K >> 6;   // K / 64

#if HAS_TDM
    if (wid == 0) {
        tdm_load_tile(lds_off(&lA[0][0]), A + (size_t)rowBase * lda, lda);
        tdm_load_tile(lds_off(&lB[0][0]), Bt + (size_t)colBase * ldb, ldb);
    }
    for (int s = 0; s < nstage; ++s) {
        if (wid == 0) {
            if (s + 1 < nstage) {
                int buf = (s + 1) & 1;
                tdm_load_tile(lds_off(&lA[buf][0]),
                              A + (size_t)rowBase * lda + (s + 1) * 64, lda);
                tdm_load_tile(lds_off(&lB[buf][0]),
                              Bt + (size_t)colBase * ldb + (s + 1) * 64, ldb);
                __builtin_amdgcn_s_wait_tensorcnt(2);   // stage s complete
            } else {
                __builtin_amdgcn_s_wait_tensorcnt(0);   // tail
            }
        }
        __syncthreads();
        compute_stage(lA[s & 1], lB[s & 1], wm, wn, acc);
        __syncthreads();
    }
#else
    for (int s = 0; s < nstage; ++s) {
        const bf16_t* gA = A + (size_t)rowBase * lda + s * 64;
        const bf16_t* gB = Bt + (size_t)colBase * ldb + s * 64;
        for (int i = tid; i < 128 * 8; i += 256) {   // 128 rows x 8 chunks of 8 elems
            int r = i >> 3, c = (i & 7) * 8;
            *(v8bf*)&lA[0][r * LDS_STRIDE + c] = *(const v8bf*)&gA[(size_t)r * lda + c];
            *(v8bf*)&lB[0][r * LDS_STRIDE + c] = *(const v8bf*)&gB[(size_t)r * ldb + c];
        }
        __syncthreads();
        compute_stage(lA[0], lB[0], wm, wn, acc);
        __syncthreads();
    }
#endif
}

// ---------------- prep kernels ----------------
__global__ void cvt_bf16_kernel(const float* __restrict__ in, bf16_t* __restrict__ out, int n) {
    int i = blockIdx.x * blockDim.x + threadIdx.x;
    if (i < n) out[i] = (bf16_t)in[i];
}

// out[n*K + k] = in[k*N + n]
__global__ void transpose_bf16_kernel(const float* __restrict__ in, bf16_t* __restrict__ out,
                                      int K, int N) {
    int i = blockIdx.x * blockDim.x + threadIdx.x;
    if (i < K * N) {
        int n = i / K, k = i % K;
        out[i] = (bf16_t)in[(size_t)k * N + n];
    }
}

__global__ void zero_bf16_kernel(bf16_t* __restrict__ p, int n8) {
    int i = blockIdx.x * blockDim.x + threadIdx.x;
    if (i < n8) { v8bf z = {}; *(v8bf*)(p + (size_t)i * 8) = z; }
}

// ---------------- 1) QKV GEMM: 128x128 blocks, TDM-fed LDS ----------------
__global__ void __launch_bounds__(256) qkv_gemm_kernel(
        const bf16_t* __restrict__ xbf, const bf16_t* __restrict__ wt,
        const float* __restrict__ bias,
        bf16_t* __restrict__ qbf, bf16_t* __restrict__ kbf, bf16_t* __restrict__ vtbf) {
    int rowBase = blockIdx.y * 128;
    int colBase = blockIdx.x * 128;
    v8f acc[2][4] = {};
    gemm128_core(xbf, wt, DIM, rowBase, colBase, DIM, DIM, acc);

    int lane = threadIdx.x & 31, hf = lane >> 4, nl = lane & 15;
    int wid = threadIdx.x >> 5, wm = wid >> 1, wn = wid & 1;
#pragma unroll
    for (int mi = 0; mi < 2; ++mi)
#pragma unroll
    for (int ni = 0; ni < 4; ++ni) {
        int n  = colBase + wn * 64 + ni * 16 + nl;
        int qi = n / DIM;            // 0=q 1=k 2=v
        int r  = n % DIM;
        int h  = r >> 6;             // head
        int d  = r & 63;             // channel
        float bv = bias[n];
#pragma unroll
        for (int j = 0; j < 8; ++j) {
            int m  = rowBase + wm * 32 + mi * 16 + j + 8 * hf;
            int bi = m / LL;
            int l  = m % LL;
            int bb = h * BATCH + bi;
            bf16_t o = (bf16_t)(acc[mi][ni][j] + bv);
            if (qi == 0)      qbf[((size_t)bb * LL + l) * HD + d] = o;
            else if (qi == 1) kbf[((size_t)bb * LL + l) * HD + d] = o;
            else              vtbf[((size_t)bb * HD + d) * KPAD + l] = o;
        }
    }
}

// ---------------- 2) decomposed rel-pos: relh[bb,l,kh], relw[bb,l,kw] ----------------
__global__ void relpos_kernel(const bf16_t* __restrict__ qbf,
                              const float* __restrict__ rph, const float* __restrict__ rpw,
                              float* __restrict__ relh, float* __restrict__ relw) {
    int idx = blockIdx.x * blockDim.x + threadIdx.x;   // (bb*196 + l)*14 + kk
    if (idx >= BB * LL * HH) return;
    int kk = idx % HH;
    int t  = idx / HH;
    int l  = t % LL;
    int bb = t / LL;
    int hq = l / WW, wq = l % WW;
    const bf16_t* q  = qbf + ((size_t)bb * LL + l) * HD;
    const float*  ph = rph + (size_t)(hq - kk + HH - 1) * HD;
    const float*  pw = rpw + (size_t)(wq - kk + WW - 1) * HD;
    float sh = 0.f, sw = 0.f;
#pragma unroll 8
    for (int c = 0; c < HD; ++c) {
        float qv = (float)q[c];
        sh += qv * ph[c];
        sw += qv * pw[c];
    }
    relh[idx] = sh;
    relw[idx] = sw;
}

// ---------------- 3) attn scores: S = (q k^T)*scale + relh + relw ----------------
__global__ void scores_kernel(const bf16_t* __restrict__ qbf, const bf16_t* __restrict__ kbf,
                              const float* __restrict__ relh, const float* __restrict__ relw,
                              float* __restrict__ attn) {
    int wave = (blockIdx.x * blockDim.x + threadIdx.x) >> 5;
    int bb = wave / (QT * QT);
    int r  = wave % (QT * QT);
    int qt = r / QT, kt = r % QT;
    const bf16_t* qb = qbf + (size_t)bb * LL * HD;
    const bf16_t* kb = kbf + (size_t)bb * LL * HD;
    v8f acc = {};
#pragma unroll
    for (int k0 = 0; k0 < HD; k0 += 32) {
        v16bf a = load_a_guard(qb, HD, qt * 16, k0, LL);
        v16bf b = load_b_guard(kb, HD, kt * 16, k0, LL);
        acc = wmma_bf16(a, b, acc);
    }
    int lane = threadIdx.x & 31, hf = lane >> 4, nl = lane & 15;
    int key = kt * 16 + nl;
    if (key < LL) {
        int kh = key / WW, kw = key % WW;
#pragma unroll
        for (int j = 0; j < 8; ++j) {
            int q = qt * 16 + j + 8 * hf;
            if (q < LL) {
                size_t rq = (size_t)bb * LL + q;
                float s = acc[j] * 0.125f + relh[rq * HH + kh] + relw[rq * HH + kw];
                attn[rq * LL + key] = s;
            }
        }
    }
}

// ---------------- 4) softmax: one wave per row, bf16 probs padded to 224 ----------------
__global__ void softmax_kernel(const float* __restrict__ attn, bf16_t* __restrict__ pbf) {
    int row  = (blockIdx.x * blockDim.x + threadIdx.x) >> 5;
    int lane = threadIdx.x & 31;
    const float* rp = attn + (size_t)row * LL;
    float v[7];
    float mx = -1e30f;
#pragma unroll
    for (int i = 0; i < 7; ++i) {
        int k = lane + i * 32;
        v[i] = (k < LL) ? rp[k] : -1e30f;
        mx = fmaxf(mx, v[i]);
    }
#pragma unroll
    for (int off = 16; off >= 1; off >>= 1)
        mx = fmaxf(mx, __shfl_xor(mx, off, 32));
    float sum = 0.f;
#pragma unroll
    for (int i = 0; i < 7; ++i) {
        int k = lane + i * 32;
        v[i] = (k < LL) ? __expf(v[i] - mx) : 0.f;
        sum += v[i];
    }
#pragma unroll
    for (int off = 16; off >= 1; off >>= 1)
        sum += __shfl_xor(sum, off, 32);
    float inv = 1.f / sum;
    bf16_t* prow = pbf + (size_t)row * KPAD;
#pragma unroll
    for (int i = 0; i < 7; ++i)          // 7*32 = 224: pad cols get zeros
        prow[lane + i * 32] = (bf16_t)(v[i] * inv);
}

// ---------------- 5) out = P @ V, scatter to (B, L, nh*hd) bf16 ----------------
__global__ void av_kernel(const bf16_t* __restrict__ pbf, const bf16_t* __restrict__ vtbf,
                          bf16_t* __restrict__ obf) {
    int wave = (blockIdx.x * blockDim.x + threadIdx.x) >> 5;
    int bb = wave / QT;
    int qt = wave % QT;
    const bf16_t* pb = pbf  + (size_t)bb * LL * KPAD;
    const bf16_t* vb = vtbf + (size_t)bb * HD * KPAD;
    v8f acc[4] = {};
    for (int k0 = 0; k0 < KPAD; k0 += 32) {
        v16bf a = load_a_guard(pb, KPAD, qt * 16, k0, LL);
#pragma unroll
        for (int s = 0; s < 4; ++s) {
            v16bf b = load_b(vb, KPAD, s * 16, k0);
            acc[s] = wmma_bf16(a, b, acc[s]);
        }
    }
    int lane = threadIdx.x & 31, hf = lane >> 4, nl = lane & 15;
    int h  = bb >> 7;
    int bi = bb & 127;
#pragma unroll
    for (int s = 0; s < 4; ++s) {
        int c = s * 16 + nl;
#pragma unroll
        for (int j = 0; j < 8; ++j) {
            int q = qt * 16 + j + 8 * hf;
            if (q < LL)
                obf[((size_t)bi * LL + q) * DIM + h * HD + c] = (bf16_t)acc[s][j];
        }
    }
}

// ---------------- 6) out projection: 128x128 blocks, TDM-fed LDS ----------------
__global__ void __launch_bounds__(256) proj_gemm_kernel(
        const bf16_t* __restrict__ obf, const bf16_t* __restrict__ wt,
        const float* __restrict__ bias, float* __restrict__ out) {
    int rowBase = blockIdx.y * 128;
    int colBase = blockIdx.x * 128;
    v8f acc[2][4] = {};
    gemm128_core(obf, wt, DIM, rowBase, colBase, DIM, DIM, acc);

    int lane = threadIdx.x & 31, hf = lane >> 4, nl = lane & 15;
    int wid = threadIdx.x >> 5, wm = wid >> 1, wn = wid & 1;
#pragma unroll
    for (int mi = 0; mi < 2; ++mi)
#pragma unroll
    for (int ni = 0; ni < 4; ++ni) {
        int n = colBase + wn * 64 + ni * 16 + nl;
        float bv = bias[n];
#pragma unroll
        for (int j = 0; j < 8; ++j) {
            int m = rowBase + wm * 32 + mi * 16 + j + 8 * hf;
            out[(size_t)m * DIM + n] = acc[mi][ni][j] + bv;
        }
    }
}

// ---------------- launch ----------------
static inline size_t align256(size_t x) { return (x + 255) & ~(size_t)255; }

extern "C" void kernel_launch(void* const* d_in, const int* in_sizes, int n_in,
                              void* d_out, int out_size, void* d_ws, size_t ws_size,
                              hipStream_t stream) {
    const float* x      = (const float*)d_in[0];   // (128,14,14,768)
    const float* qkv_w  = (const float*)d_in[1];   // (768, 2304)
    const float* qkv_b  = (const float*)d_in[2];   // (2304,)
    const float* proj_w = (const float*)d_in[3];   // (768, 768)
    const float* proj_b = (const float*)d_in[4];   // (768,)
    const float* rph    = (const float*)d_in[5];   // (27, 64)
    const float* rpw    = (const float*)d_in[6];   // (27, 64)
    float* out = (float*)d_out;                    // (128,14,14,768)

    char* ws = (char*)d_ws;
    size_t off = 0;
    auto take = [&](size_t bytes) { char* p = ws + off; off += align256(bytes); return p; };
    bf16_t* xbf   = (bf16_t*)take((size_t)MROWS * DIM * 2);
    bf16_t* wqkvt = (bf16_t*)take((size_t)N3 * DIM * 2);
    bf16_t* wprjt = (bf16_t*)take((size_t)DIM * DIM * 2);
    bf16_t* qbf   = (bf16_t*)take((size_t)BB * LL * HD * 2);
    bf16_t* kbf   = (bf16_t*)take((size_t)BB * LL * HD * 2);
    bf16_t* vtbf  = (bf16_t*)take((size_t)BB * HD * KPAD * 2);
    float*  relh  = (float*) take((size_t)BB * LL * HH * 4);
    float*  relw  = (float*) take((size_t)BB * LL * HH * 4);
    float*  attn  = (float*) take((size_t)BB * LL * LL * 4);
    bf16_t* pbf   = (bf16_t*)take((size_t)BB * LL * KPAD * 2);
    bf16_t* obf   = (bf16_t*)take((size_t)MROWS * DIM * 2);
    (void)ws_size;

    const int TB = 256;

    { int n = MROWS * DIM;
      cvt_bf16_kernel<<<(n + TB - 1) / TB, TB, 0, stream>>>(x, xbf, n); }
    { int n = N3 * DIM;
      transpose_bf16_kernel<<<(n + TB - 1) / TB, TB, 0, stream>>>(qkv_w, wqkvt, DIM, N3); }
    { int n = DIM * DIM;
      transpose_bf16_kernel<<<(n + TB - 1) / TB, TB, 0, stream>>>(proj_w, wprjt, DIM, DIM); }
    { int n8 = (BB * HD * KPAD) / 8;
      zero_bf16_kernel<<<(n8 + TB - 1) / TB, TB, 0, stream>>>(vtbf, n8); }

    // 1) QKV GEMM: 196 x 18 blocks of 128x128 (exact)
    qkv_gemm_kernel<<<dim3(N3 / 128, MROWS / 128), TB, 0, stream>>>(
        xbf, wqkvt, qkv_b, qbf, kbf, vtbf);

    // 2) rel-pos einsums
    relpos_kernel<<<(BB * LL * HH) / TB, TB, 0, stream>>>(qbf, rph, rpw, relh, relw);

    // 3) scores: 1536*13*13 waves (exact)
    scores_kernel<<<(BB * QT * QT) / 8, TB, 0, stream>>>(qbf, kbf, relh, relw, attn);

    // 4) softmax: one wave per row (exact)
    softmax_kernel<<<(BB * LL) / 8, TB, 0, stream>>>(attn, pbf);

    // 5) attn @ V: 1536*13 waves (exact)
    av_kernel<<<(BB * QT) / 8, TB, 0, stream>>>(pbf, vtbf, obf);

    // 6) out projection: 196 x 6 blocks of 128x128 (exact)
    proj_gemm_kernel<<<dim3(DIM / 128, MROWS / 128), TB, 0, stream>>>(
        obf, wprjt, proj_b, out);
}
